// CausalSelfAttentionRelPos_74844100100434
// MI455X (gfx1250) — matrix operand order repaired and checked by reading, not compiled, not run on hardware
//
#include <hip/hip_runtime.h>
#include <hip/hip_bf16.h>
#include <stdint.h>

// ---------------------------------------------------------------------------
// CDNA5 (gfx1250) wave32 WMMA implementation of causal self-attention with
// Music-Transformer relative position embeddings.
//   B=2, L=2048, D=1024, NH=16, HS=64, BLOCK_SIZE=2048 (start offset == 0)
// All GEMMs run on V_WMMA_F32_16X16X32_BF16 (bf16 operands, f32 accum).
//
// GEMM kernels: block tile 128x64, BK=64, 8 waves each owning a 32x32 output
// (2x2 C fragments) -> 8 WMMAs per K-step per wave between barriers.
// Attention: flash-style, 32-key steps, 14 WMMAs per step per wave, Q held
// in registers; K/Er streamed from L2, V transposed through LDS.
// ---------------------------------------------------------------------------

#define L_SEQ   2048
#define D_MODEL 1024
#define NHEAD   16
#define HS      64
#define BATCH   2

typedef __attribute__((ext_vector_type(16))) __bf16 v16bf;
typedef __attribute__((ext_vector_type(8)))  float  v8f;

// Native bf16 convert (v_cvt_*_bf16_f32 on gfx1250), RNE.
__device__ __forceinline__ unsigned short f32_to_bf16(float f) {
  union { __bf16 b; unsigned short u; } c;
  c.b = (__bf16)f;
  return c.u;
}
__device__ __forceinline__ unsigned pack_bf2(float a, float b) {
  union { __bf16 b[2]; unsigned u; } c;
  c.b[0] = (__bf16)a;
  c.b[1] = (__bf16)b;
  return c.u;
}

union ABFrag { v16bf v; uint4 q[2]; };

// A fragment (16x32, M x K) from a row-major bf16 matrix.
// Layout: lanes 0-15 row=lane, K 0..15; lanes 16-31 row=lane-16,
// K 16..31; each chunk is 8 contiguous bf16 = one 16B load.
__device__ __forceinline__ v16bf load_a_rowmajor(const unsigned short* base,
                                                 int pitch, int lane, int kk) {
  ABFrag f;
  const int grp = lane >> 4;
  const int r   = lane & 15;
  const unsigned short* row = base + (long)r * pitch + kk;
  f.q[0] = *(const uint4*)(row + grp * 16);
  f.q[1] = *(const uint4*)(row + grp * 16 + 8);
  return f.v;
}

// B fragment (32x16, K x N) where the SOURCE is stored N x K row-major
// (i.e. "keys as rows"). Lanes 0-15: n=lane, K 0..15 (32 contiguous bytes);
// lanes 16-31: n=lane-16, K 16..31.
__device__ __forceinline__ v16bf load_b_nk(const unsigned short* base,
                                           int pitch, int lane, int kk) {
  ABFrag f;
  const int grp = lane >> 4;
  const int n   = lane & 15;
  const unsigned short* row = base + (long)n * pitch + kk + grp * 16;
  f.q[0] = *(const uint4*)(row);
  f.q[1] = *(const uint4*)(row + 8);
  return f.v;
}

__device__ __forceinline__ v8f wmma_bf16(v16bf a, v16bf b, v8f c) {
  return __builtin_amdgcn_wmma_f32_16x16x32_bf16(false, a, false, b,
                                                 (short)0, c, false, false);
}

// ---------------------------------------------------------------------------
// Kernel 1: Er (f32) -> bf16
// ---------------------------------------------------------------------------
__global__ void relpos_er_cvt_kernel(const float* __restrict__ er,
                                     unsigned short* __restrict__ erb, int n) {
  for (int i = blockIdx.x * blockDim.x + threadIdx.x; i < n;
       i += gridDim.x * blockDim.x)
    erb[i] = f32_to_bf16(er[i]);
}

// ---------------------------------------------------------------------------
// Kernel 2: qkv = x @ Wqkv + bqkv   (M=4096, N=3072, K=1024)
// Block tile 128x64, BK=64; 8 waves = 4(M) x 2(N), each wave 32x32 output.
// Output scattered as bf16 into head-major q/k/v: [B, NH, L, HS].
// ---------------------------------------------------------------------------
__global__ __launch_bounds__(256)
void relpos_qkv_gemm_kernel(const float* __restrict__ x,
                            const float* __restrict__ Wqkv,
                            const float* __restrict__ bqkv,
                            unsigned short* __restrict__ qb,
                            unsigned short* __restrict__ kb,
                            unsigned short* __restrict__ vb) {
  __shared__ unsigned short As[128 * 64];  // x tile, M x K row-major (16 KB)
  __shared__ unsigned short Bs[64 * 64];   // W tile transposed, N x K (8 KB)

  const int tid  = threadIdx.x;
  const int lane = tid & 31;
  const int w    = tid >> 5;
  const int mw   = w & 3;       // 32-row group 0..3
  const int nw   = w >> 2;      // 32-col group 0..1
  const int m0   = blockIdx.x * 128;
  const int n0   = blockIdx.y * 64;

  v8f c00, c01, c10, c11;
  #pragma unroll
  for (int v = 0; v < 8; ++v) { c00[v] = c01[v] = c10[v] = c11[v] = 0.f; }

  const int arow = tid >> 1, ac = (tid & 1) * 32;
  const int bk   = tid & 63, bn16 = (tid >> 6) * 16;

  for (int k0 = 0; k0 < D_MODEL; k0 += 64) {
    {  // stage x -> bf16 LDS: 32 contiguous f32 per thread -> 4x 16B stores
      const float* src = x + (long)(m0 + arow) * D_MODEL + k0 + ac;
      #pragma unroll
      for (int j2 = 0; j2 < 4; ++j2) {
        uint4 pk;
        pk.x = pack_bf2(src[8 * j2 + 0], src[8 * j2 + 1]);
        pk.y = pack_bf2(src[8 * j2 + 2], src[8 * j2 + 3]);
        pk.z = pack_bf2(src[8 * j2 + 4], src[8 * j2 + 5]);
        pk.w = pack_bf2(src[8 * j2 + 6], src[8 * j2 + 7]);
        *(uint4*)(&As[arow * 64 + ac + 8 * j2]) = pk;
      }
    }
    {  // stage Wqkv transposed -> Bs[n][k]
      const float* src = Wqkv + (long)(k0 + bk) * 3072 + n0 + bn16;
      #pragma unroll
      for (int j = 0; j < 16; ++j)
        Bs[(bn16 + j) * 64 + bk] = f32_to_bf16(src[j]);
    }
    __syncthreads();
    #pragma unroll
    for (int kk = 0; kk < 64; kk += 32) {
      v16bf a0 = load_a_rowmajor(As + (mw * 32 + 0)  * 64, 64, lane, kk);
      v16bf a1 = load_a_rowmajor(As + (mw * 32 + 16) * 64, 64, lane, kk);
      v16bf b0 = load_b_nk(Bs + (nw * 32 + 0)  * 64, 64, lane, kk);
      v16bf b1 = load_b_nk(Bs + (nw * 32 + 16) * 64, 64, lane, kk);
      c00 = wmma_bf16(a0, b0, c00);
      c01 = wmma_bf16(a0, b1, c01);
      c10 = wmma_bf16(a1, b0, c10);
      c11 = wmma_bf16(a1, b1, c11);
    }
    __syncthreads();
  }

  const int grp = lane >> 4, ln = lane & 15;
  #pragma unroll
  for (int tn = 0; tn < 2; ++tn) {
    const int n_g  = n0 + nw * 32 + tn * 16 + ln;
    const float bias = bqkv[n_g];
    const int mat  = n_g >> 10;          // 0:q 1:k 2:v
    const int dd   = n_g & 1023;
    const int head = dd >> 6, hs = dd & 63;
    unsigned short* dst = (mat == 0) ? qb : (mat == 1) ? kb : vb;
    #pragma unroll
    for (int tm = 0; tm < 2; ++tm) {
      v8f& c = tm ? (tn ? c11 : c10) : (tn ? c01 : c00);
      #pragma unroll
      for (int v = 0; v < 8; ++v) {
        const int ig = m0 + mw * 32 + tm * 16 + grp * 8 + v;  // b*L + l
        const int bbi = ig >> 11, l = ig & (L_SEQ - 1);
        const long idx = (((long)(bbi * NHEAD + head) * L_SEQ + l) << 6) + hs;
        dst[idx] = f32_to_bf16(c[v] + bias);
      }
    }
  }
}

// ---------------------------------------------------------------------------
// Kernel 3: flash attention with rel-pos skew.
// One block per (b, head, 128-row query tile); 8 waves, 16 query rows each.
// Key step = 32. Scores: S = (q.k^T + q.Er[m+L-1-l]^T) * scale, causal mask,
// online softmax, P@V accumulated in f32 fragments.
// ---------------------------------------------------------------------------
__global__ __launch_bounds__(256)
void relpos_flash_attn_kernel(const unsigned short* __restrict__ qb,
                              const unsigned short* __restrict__ kb,
                              const unsigned short* __restrict__ vb,
                              const unsigned short* __restrict__ erb,
                              unsigned short* __restrict__ yb) {
  __shared__ unsigned short Vt[HS * 32];        // V tile transposed [hs][key]
  __shared__ float          Uscr[8][16 * 48];   // per-wave rel band scores
  __shared__ unsigned short Pscr[8][16 * 32];   // per-wave P tile (bf16)

  const int tid  = threadIdx.x;
  const int lane = tid & 31;
  const int w    = tid >> 5;
  const int grp  = lane >> 4, ln = lane & 15;

  const int bidx  = blockIdx.x;
  const int lt    = bidx & 15;
  const int hh    = (bidx >> 4) & 15;
  const int bb    = bidx >> 8;
  const int l0blk = lt * 128;
  const int l0w   = l0blk + w * 16;

  const long head_off = (long)(bb * NHEAD + hh) * L_SEQ * HS;
  const unsigned short* qh = qb + head_off;
  const unsigned short* kh = kb + head_off;
  const unsigned short* vh = vb + head_off;

  // Q fragments (16 rows x K=64) held in registers for the whole kernel.
  const v16bf qa0 = load_a_rowmajor(qh + (long)l0w * HS, HS, lane, 0);
  const v16bf qa1 = load_a_rowmajor(qh + (long)l0w * HS, HS, lane, 32);

  v8f o0, o1, o2, o3;
  float vmax[8], vsum[8];
  #pragma unroll
  for (int v = 0; v < 8; ++v) {
    o0[v] = o1[v] = o2[v] = o3[v] = 0.f;
    vmax[v] = -3.0e38f;
    vsum[v] = 0.f;
  }

  const float scale = 0.125f;                 // 1/sqrt(HS)
  const int sr = tid & 31, sh8 = (tid >> 5) * 8;
  const int mend = l0blk + 128;
  float*          uw = &Uscr[w][0];
  unsigned short* pw = &Pscr[w][0];

  for (int m0 = 0; m0 < mend; m0 += 32) {
    __syncthreads();
    {  // stage V tile transposed into LDS: Vt[hs][key]
      const unsigned short* src = vh + (long)(m0 + sr) * HS + sh8;
      #pragma unroll
      for (int j = 0; j < 8; ++j) Vt[(sh8 + j) * 32 + sr] = src[j];
    }
    if (m0 + 32 < mend) {  // global_prefetch_b8 for next K/V tiles (L2 warm)
      __builtin_prefetch(vh + (long)(m0 + 32 + sr) * HS + sh8, 0, 1);
      __builtin_prefetch(kh + (long)(m0 + 32 + sr) * HS + sh8, 0, 1);
    }
    __syncthreads();

    // ---- QK^T: two 16-column score tiles, streamed from L2 --------------
    v8f s0, s1;
    #pragma unroll
    for (int v = 0; v < 8; ++v) { s0[v] = 0.f; s1[v] = 0.f; }
    {
      v16bf k0a = load_b_nk(kh + (long)(m0)      * HS, HS, lane, 0);
      v16bf k0b = load_b_nk(kh + (long)(m0)      * HS, HS, lane, 32);
      v16bf k1a = load_b_nk(kh + (long)(m0 + 16) * HS, HS, lane, 0);
      v16bf k1b = load_b_nk(kh + (long)(m0 + 16) * HS, HS, lane, 32);
      s0 = wmma_bf16(qa0, k0a, s0);
      s0 = wmma_bf16(qa1, k0b, s0);
      s1 = wmma_bf16(qa0, k1a, s1);
      s1 = wmma_bf16(qa1, k1b, s1);
    }

    // ---- rel-pos band GEMM: U[i][jj] = q_i . Er[base + jj], jj in [0,48) --
    const int base = m0 + (L_SEQ - 1) - l0w - 15;
    v8f u0, u1, u2;
    #pragma unroll
    for (int v = 0; v < 8; ++v) { u0[v] = 0.f; u1[v] = 0.f; u2[v] = 0.f; }
    #pragma unroll
    for (int f = 0; f < 3; ++f) {
      int r = base + f * 16 + ln;
      r = r < 0 ? 0 : (r > L_SEQ - 1 ? L_SEQ - 1 : r);   // clamped band edge
      const unsigned short* row = erb + (long)r * HS;
      ABFrag fb0, fb1;
      fb0.q[0] = *(const uint4*)(row + grp * 16);
      fb0.q[1] = *(const uint4*)(row + grp * 16 + 8);
      fb1.q[0] = *(const uint4*)(row + 32 + grp * 16);
      fb1.q[1] = *(const uint4*)(row + 32 + grp * 16 + 8);
      v8f& u = (f == 0) ? u0 : ((f == 1) ? u1 : u2);
      u = wmma_bf16(qa0, fb0.v, u);
      u = wmma_bf16(qa1, fb1.v, u);
    }

    // Spill U to per-wave LDS, then gather the skew diagonal U[i][j+15-i].
    #pragma unroll
    for (int v = 0; v < 8; ++v) {
      const int i = grp * 8 + v;
      uw[i * 48 + 0  + ln] = u0[v];
      uw[i * 48 + 16 + ln] = u1[v];
      uw[i * 48 + 32 + ln] = u2[v];
    }
    float sv0[8], sv1[8];
    #pragma unroll
    for (int v = 0; v < 8; ++v) {
      const int i  = grp * 8 + v;
      const int j0 = ln, j1 = 16 + ln;
      const float rel0 = uw[i * 48 + (j0 + 15 - i)];
      const float rel1 = uw[i * 48 + (j1 + 15 - i)];
      float t0 = (s0[v] + rel0) * scale;
      float t1 = (s1[v] + rel1) * scale;
      if (m0 + j0 > l0w + i) t0 = -3.0e38f;   // causal mask
      if (m0 + j1 > l0w + i) t1 = -3.0e38f;
      sv0[v] = t0; sv1[v] = t1;
    }

    // ---- online softmax (rows live per-VGPR across 16-lane groups) -------
    #pragma unroll
    for (int v = 0; v < 8; ++v) {
      float m = fmaxf(sv0[v], sv1[v]);
      m = fmaxf(m, __shfl_xor(m, 1, 32));
      m = fmaxf(m, __shfl_xor(m, 2, 32));
      m = fmaxf(m, __shfl_xor(m, 4, 32));
      m = fmaxf(m, __shfl_xor(m, 8, 32));
      const float nm   = fmaxf(vmax[v], m);
      const float corr = __expf(vmax[v] - nm);
      const float p0 = __expf(sv0[v] - nm);
      const float p1 = __expf(sv1[v] - nm);
      float ps = p0 + p1;
      ps += __shfl_xor(ps, 1, 32);
      ps += __shfl_xor(ps, 2, 32);
      ps += __shfl_xor(ps, 4, 32);
      ps += __shfl_xor(ps, 8, 32);
      vsum[v] = vsum[v] * corr + ps;
      vmax[v] = nm;
      o0[v] *= corr; o1[v] *= corr; o2[v] *= corr; o3[v] *= corr;
      const int i = grp * 8 + v;
      pw[i * 32 + ln]      = f32_to_bf16(p0);
      pw[i * 32 + 16 + ln] = f32_to_bf16(p1);
    }

    // ---- P @ V (K=32 keys in one WMMA step, N=64 in 4 tiles) -------------
    v16bf pa  = load_a_rowmajor(pw, 32, lane, 0);
    v16bf bv0 = load_b_nk(&Vt[0 * 16 * 32], 32, lane, 0);
    v16bf bv1 = load_b_nk(&Vt[1 * 16 * 32], 32, lane, 0);
    v16bf bv2 = load_b_nk(&Vt[2 * 16 * 32], 32, lane, 0);
    v16bf bv3 = load_b_nk(&Vt[3 * 16 * 32], 32, lane, 0);
    o0 = wmma_bf16(pa, bv0, o0);
    o1 = wmma_bf16(pa, bv1, o1);
    o2 = wmma_bf16(pa, bv2, o2);
    o3 = wmma_bf16(pa, bv3, o3);
  }

  // epilogue: y[b][l][h*64 + j] = o / rowsum  (bf16 for the proj GEMM)
  #pragma unroll
  for (int v = 0; v < 8; ++v) {
    const float inv = 1.f / vsum[v];
    const int i = grp * 8 + v;
    const long rowbase = (long)(bb * L_SEQ + l0w + i) * D_MODEL + hh * HS;
    yb[rowbase + 0  + ln] = f32_to_bf16(o0[v] * inv);
    yb[rowbase + 16 + ln] = f32_to_bf16(o1[v] * inv);
    yb[rowbase + 32 + ln] = f32_to_bf16(o2[v] * inv);
    yb[rowbase + 48 + ln] = f32_to_bf16(o3[v] * inv);
  }
}

// ---------------------------------------------------------------------------
// Kernel 4: out = y @ Wproj + bproj   (M=4096, N=1024, K=1024), f32 output.
// Same 128x64xBK64 tiling as kernel 2; A operand is already bf16.
// ---------------------------------------------------------------------------
__global__ __launch_bounds__(256)
void relpos_proj_gemm_kernel(const unsigned short* __restrict__ yb,
                             const float* __restrict__ Wp,
                             const float* __restrict__ bp,
                             float* __restrict__ out) {
  __shared__ unsigned short As[128 * 64];
  __shared__ unsigned short Bs[64 * 64];

  const int tid  = threadIdx.x;
  const int lane = tid & 31;
  const int w    = tid >> 5;
  const int mw   = w & 3;
  const int nw   = w >> 2;
  const int m0   = blockIdx.x * 128;
  const int n0   = blockIdx.y * 64;

  v8f c00, c01, c10, c11;
  #pragma unroll
  for (int v = 0; v < 8; ++v) { c00[v] = c01[v] = c10[v] = c11[v] = 0.f; }

  const int arow = tid >> 1, ac = (tid & 1) * 32;
  const int bk   = tid & 63, bn16 = (tid >> 6) * 16;

  for (int k0 = 0; k0 < D_MODEL; k0 += 64) {
    {  // stage y (already bf16): 64B per thread = 4x 16B copies
      const unsigned short* src = yb + (long)(m0 + arow) * D_MODEL + k0 + ac;
      #pragma unroll
      for (int j2 = 0; j2 < 4; ++j2)
        *(uint4*)(&As[arow * 64 + ac + 8 * j2]) =
            *(const uint4*)(src + 8 * j2);
    }
    {  // stage Wproj transposed -> Bs[n][k]
      const float* src = Wp + (long)(k0 + bk) * D_MODEL + n0 + bn16;
      #pragma unroll
      for (int j = 0; j < 16; ++j)
        Bs[(bn16 + j) * 64 + bk] = f32_to_bf16(src[j]);
    }
    __syncthreads();
    #pragma unroll
    for (int kk = 0; kk < 64; kk += 32) {
      v16bf a0 = load_a_rowmajor(As + (mw * 32 + 0)  * 64, 64, lane, kk);
      v16bf a1 = load_a_rowmajor(As + (mw * 32 + 16) * 64, 64, lane, kk);
      v16bf b0 = load_b_nk(Bs + (nw * 32 + 0)  * 64, 64, lane, kk);
      v16bf b1 = load_b_nk(Bs + (nw * 32 + 16) * 64, 64, lane, kk);
      c00 = wmma_bf16(a0, b0, c00);
      c01 = wmma_bf16(a0, b1, c01);
      c10 = wmma_bf16(a1, b0, c10);
      c11 = wmma_bf16(a1, b1, c11);
    }
    __syncthreads();
  }

  const int grp = lane >> 4, ln = lane & 15;
  #pragma unroll
  for (int tn = 0; tn < 2; ++tn) {
    const int n_g = n0 + nw * 32 + tn * 16 + ln;
    const float bias = bp[n_g];
    #pragma unroll
    for (int tm = 0; tm < 2; ++tm) {
      v8f& c = tm ? (tn ? c11 : c10) : (tn ? c01 : c00);
      #pragma unroll
      for (int v = 0; v < 8; ++v) {
        const int ig = m0 + mw * 32 + tm * 16 + grp * 8 + v;
        out[(long)ig * D_MODEL + n_g] = c[v] + bias;
      }
    }
  }
}

// ---------------------------------------------------------------------------
extern "C" void kernel_launch(void* const* d_in, const int* in_sizes, int n_in,
                              void* d_out, int out_size, void* d_ws, size_t ws_size,
                              hipStream_t stream) {
  const float* x     = (const float*)d_in[0];
  const float* Wqkv  = (const float*)d_in[1];
  const float* bqkv  = (const float*)d_in[2];
  const float* Wproj = (const float*)d_in[3];
  const float* bproj = (const float*)d_in[4];
  const float* Er    = (const float*)d_in[5];
  float* out = (float*)d_out;

  const size_t per_qkv = (size_t)BATCH * NHEAD * L_SEQ * HS;  // 4M bf16
  unsigned short* qb  = (unsigned short*)d_ws;
  unsigned short* kb  = qb + per_qkv;
  unsigned short* vb  = kb + per_qkv;
  unsigned short* erb = vb + per_qkv;
  unsigned short* yb  = erb + (size_t)L_SEQ * HS;

  relpos_er_cvt_kernel<<<128, 256, 0, stream>>>(Er, erb, L_SEQ * HS);
  relpos_qkv_gemm_kernel<<<dim3(32, 48), 256, 0, stream>>>(x, Wqkv, bqkv,
                                                           qb, kb, vb);
  relpos_flash_attn_kernel<<<BATCH * NHEAD * (L_SEQ / 128), 256, 0, stream>>>(
      qb, kb, vb, erb, yb);
  relpos_proj_gemm_kernel<<<dim3(32, 16), 256, 0, stream>>>(yb, Wproj, bproj,
                                                            out);
}
